// GAT_58832462020773
// MI455X (gfx1250) — matrix operand order repaired and checked by reading, not compiled, not run on hardware
//
#include <hip/hip_runtime.h>
#include <hip/hip_bf16.h>

typedef __attribute__((ext_vector_type(16))) _Float16 v16h;
typedef __attribute__((ext_vector_type(8)))  float    v8f;

// N=256, H=256, DDE=64, NH=4

__device__ __forceinline__ float wred32(float v) {
  #pragma unroll
  for (int o = 16; o > 0; o >>= 1) v += __shfl_xor(v, o, 32);
  return v;
}

// ---------------- prep: tanh(ah), u1/u2 = wm@a halves, p/q dist proj, B-fragment pack ----
__global__ __launch_bounds__(256) void gat_prep(
    const float* __restrict__ ah, const float* __restrict__ goal,
    const float* __restrict__ action, const float* __restrict__ Wd,
    const float* __restrict__ Wg, const float* __restrict__ a_in,
    const float* __restrict__ wm,
    float* __restrict__ tanh_ah, float* __restrict__ u1, float* __restrict__ u2,
    float* __restrict__ p, float* __restrict__ q, _Float16* __restrict__ bpack)
{
  const int gid = blockIdx.x * 256 + threadIdx.x;   // 0..65535

  // tanh of action hidden state (N*H = 65536)
  tanh_ah[gid] = tanhf(ah[gid]);

  // u1[z,h] = wm[z,h,:]@a[:256] ; u2[z,h] = wm[z,h,:]@a[256:]
  if (gid < 2048) {
    const int sel = gid >> 10;            // 0->u1, 1->u2
    const int zh  = gid & 1023;           // z*256+h
    const float* av   = a_in + sel * 256;
    const float* wrow = wm + (size_t)zh * 256;  // (z*256+h)*256 == z*65536+h*256
    float acc = 0.f;
    for (int f = 0; f < 256; ++f) acc += wrow[f] * av[f];
    (sel ? u2 : u1)[zh] = acc;
  }

  // p[i,c] from W_dist rows 0..3, q[j,c] from rows 4..7
  if (gid < 32768) {
    const int sel = gid >> 14;            // 0->p, 1->q
    const int nc  = gid & 16383;
    const int n = nc >> 6, c = nc & 63;
    const float* wd = Wd + sel * 4 * 64;
    float v = action[n*2+0]*wd[0*64+c] + action[n*2+1]*wd[1*64+c]
            + goal[n*2+0]*wd[2*64+c]   + goal[n*2+1]*wd[3*64+c];
    (sel ? q : p)[nc] = v;
  }

  // Pre-swizzle W_gate (64x256 f32) into f16 WMMA B fragments.
  // b frag for (ft,kk): lane ln, half hh  ->  Wg[K=kk*32+(hh&7)+8*(ln>>4)+16*(hh>>3)][N=ft*16+(ln&15)]
  if (gid < 16384) {
    const int o   = gid;
    const int kt  = o >> 9;               // ft*2+kk
    const int rem = o & 511;
    const int ln  = rem >> 4;             // lane 0..31
    const int hh  = rem & 15;
    const int ft  = kt >> 1, kk = kt & 1;
    const int K   = kk*32 + (hh & 7) + 8*(ln >> 4) + 16*(hh >> 3);
    const int Nn  = ft*16 + (ln & 15);
    bpack[o] = (_Float16)Wg[K*256 + Nn];
  }
}

// ---------------- main: one block per row n, 4 waves = 4 heads, online softmax --------
__global__ __launch_bounds__(128) void gat_main(
    const float* __restrict__ ah, const float* __restrict__ gh,
    const float* __restrict__ b_dist, const float* __restrict__ b_gate,
    const float* __restrict__ wm, const float* __restrict__ bias,
    const float* __restrict__ tanh_ah,
    const float* __restrict__ u1, const float* __restrict__ u2,
    const float* __restrict__ p, const float* __restrict__ q,
    const _Float16* __restrict__ bpack,
    float* __restrict__ out)
{
  __shared__ _Float16 t_lds[16*66];     // hidden tile, f16, padded stride 66
  __shared__ float ga_lds[16*260];      // gate_action tile, padded stride 260
  __shared__ float hbar[4*256];
  __shared__ float p_s[64], bd_s[64];

  const int n    = blockIdx.x;
  const int tid  = threadIdx.x;
  const int wid  = tid >> 5;            // head z
  const int lane = tid & 31;
  const int hi   = lane >> 4;
  const int lm   = lane & 15;

  if (tid < 64) { p_s[tid] = p[n*64 + tid]; bd_s[tid] = b_dist[tid]; }

  // Per-lane state: h = r*32 + lane layout over H=256
  float u1f[8], u2f[8], ahr[8], ghr[8], acc[8];
  #pragma unroll
  for (int r = 0; r < 8; ++r) {
    const int h = r*32 + lane;
    u1f[r] = u1[wid*256 + h];
    u2f[r] = u2[wid*256 + h];
    ahr[r] = ah[n*256 + h];
    ghr[r] = gh[n*256 + h];
  }
  float dA = 0.f, d0 = 0.f, d1 = 0.f;
  #pragma unroll
  for (int r = 0; r < 8; ++r) { dA += ahr[r]*u1f[r]; d0 += ahr[r]*u2f[r]; d1 += ghr[r]*u2f[r]; }
  const float sA = wred32(dA);          // hsp@a1 term (softmax-relevant via leaky-relu)
  d0 = sA + wred32(d0);                 // score for k=0  (h = ah[n])
  d1 = sA + wred32(d1);                 // score for k=n+1 (h = gh[n])
  d0 = d0 > 0.f ? d0 : 0.2f*d0;
  d1 = d1 > 0.f ? d1 : 0.2f*d1;
  float m  = fmaxf(d0, d1);
  const float w0 = expf(d0 - m), w1 = expf(d1 - m);
  float s = w0 + w1;
  #pragma unroll
  for (int r = 0; r < 8; ++r) acc[r] = w0*ahr[r] + w1*ghr[r];

  __syncthreads();

  for (int jt = 0; jt < 16; ++jt) {
    const int j0 = jt*16;

    // Phase A: t[jj,c] = relu(p[n,c] + q[j0+jj,c] + b_dist[c]), stored f16
    #pragma unroll
    for (int i = 0; i < 8; ++i) {
      const int lin = i*128 + tid;
      const int jj = lin >> 6, c = lin & 63;
      const float tv = p_s[c] + q[(j0+jj)*64 + c] + bd_s[c];
      t_lds[jj*66 + c] = (_Float16)fmaxf(tv, 0.f);
    }
    __syncthreads();

    // Phase B: gate tile via WMMA:  sigmoid(t(16x64) @ Wg(64x256) + bg) * tanh_ah
    // A frag (16-bit A 16x32 layout): VGPR p holds K = 2p (+8 if p>=4) + 8*hi, row M = lm
    v16h a0, a1;
    #pragma unroll
    for (int pp = 0; pp < 8; ++pp) {
      const int K0 = 2*pp + (pp >= 4 ? 8 : 0) + 8*hi;
      a0[2*pp]   = t_lds[lm*66 + K0];
      a0[2*pp+1] = t_lds[lm*66 + K0 + 1];
      a1[2*pp]   = t_lds[lm*66 + 32 + K0];
      a1[2*pp+1] = t_lds[lm*66 + 32 + K0 + 1];
    }
    #pragma unroll
    for (int tix = 0; tix < 4; ++tix) {
      const int ft = wid*4 + tix;
      const int f0 = ft*16;
      const float bg = b_gate[f0 + lm];
      v8f c8;
      #pragma unroll
      for (int g = 0; g < 8; ++g) c8[g] = bg;
      const v16h b0 = *reinterpret_cast<const v16h*>(bpack + (ft*2+0)*512 + lane*16);
      const v16h b1 = *reinterpret_cast<const v16h*>(bpack + (ft*2+1)*512 + lane*16);
      c8 = __builtin_amdgcn_wmma_f32_16x16x32_f16(false, a0, false, b0, (short)0, c8, false, false);
      c8 = __builtin_amdgcn_wmma_f32_16x16x32_f16(false, a1, false, b1, (short)0, c8, false, false);
      #pragma unroll
      for (int g = 0; g < 8; ++g) {
        const int Mrow = g + 8*hi;               // C/D layout: row = g + 8*hi, col = lm
        const int j = j0 + Mrow;
        const float gate = 1.f / (1.f + expf(-c8[g]));
        ga_lds[Mrow*260 + f0 + lm] = gate * tanh_ah[j*256 + f0 + lm];
      }
    }
    __syncthreads();

    // Phase C: per-head scores + online softmax + accumulator update
    float ev[16];
    float emax = -3.0e38f;
    #pragma unroll
    for (int M = 0; M < 16; ++M) {
      float d = 0.f;
      #pragma unroll
      for (int r = 0; r < 8; ++r) d += ga_lds[M*260 + r*32 + lane] * u2f[r];
      float sc = sA + wred32(d);
      sc = sc > 0.f ? sc : 0.2f*sc;              // leaky relu
      sc = (j0 + M == n) ? -1.0e30f : sc;        // diagonal handled at init (k=n+1)
      ev[M] = sc;
      emax = fmaxf(emax, sc);
    }
    const float mnew  = fmaxf(m, emax);
    const float scale = expf(m - mnew);
    float ssum = 0.f;
    #pragma unroll
    for (int M = 0; M < 16; ++M) { ev[M] = expf(ev[M] - mnew); ssum += ev[M]; }
    s = s*scale + ssum;
    m = mnew;
    #pragma unroll
    for (int r = 0; r < 8; ++r) acc[r] *= scale;
    #pragma unroll
    for (int M = 0; M < 16; ++M) {
      #pragma unroll
      for (int r = 0; r < 8; ++r)
        acc[r] += ev[M] * ga_lds[M*260 + r*32 + lane];
    }
    __syncthreads();
  }

  const float inv = 1.f / s;
  #pragma unroll
  for (int r = 0; r < 8; ++r) hbar[wid*256 + r*32 + lane] = acc[r] * inv;
  __syncthreads();

  // Stage 2: out[n,f] = relu(0.25 * sum_{z,h} hbar[z,h]*wm[z,h,f]) + bias[f]
  #pragma unroll
  for (int half_ = 0; half_ < 2; ++half_) {
    const int f = tid + half_*128;
    float o = 0.f;
    for (int zh = 0; zh < 1024; ++zh)
      o += hbar[zh] * wm[(size_t)zh*256 + f];
    out[n*256 + f] = fmaxf(0.25f*o, 0.f) + bias[f];
  }
}

extern "C" void kernel_launch(void* const* d_in, const int* in_sizes, int n_in,
                              void* d_out, int out_size, void* d_ws, size_t ws_size,
                              hipStream_t stream) {
  const float* ah     = (const float*)d_in[0];   // action_hidden_state (256,256)
  const float* gh     = (const float*)d_in[1];   // goal_hidden_state   (256,256)
  const float* goal   = (const float*)d_in[2];   // (256,2)
  const float* action = (const float*)d_in[3];   // (256,2)
  const float* Wd     = (const float*)d_in[4];   // (8,64)
  const float* bd     = (const float*)d_in[5];   // (64,)
  const float* Wg     = (const float*)d_in[6];   // (64,256)
  const float* bg     = (const float*)d_in[7];   // (256,)
  const float* wm     = (const float*)d_in[8];   // (4,1,256,256)
  const float* a_in   = (const float*)d_in[9];   // (1,1,512,1)
  const float* bias   = (const float*)d_in[10];  // (256,)

  char* ws = (char*)d_ws;
  float*    tanh_ah = (float*)(ws + 0);          // 256 KB
  float*    u1      = (float*)(ws + 262144);     // 4 KB
  float*    u2      = (float*)(ws + 266240);     // 4 KB
  float*    p       = (float*)(ws + 270336);     // 64 KB
  float*    q       = (float*)(ws + 335872);     // 64 KB
  _Float16* bpack   = (_Float16*)(ws + 401408);  // 32 KB (32B aligned)

  gat_prep<<<256, 256, 0, stream>>>(ah, goal, action, Wd, Wg, a_in, wm,
                                    tanh_ah, u1, u2, p, q, bpack);
  gat_main<<<256, 128, 0, stream>>>(ah, gh, bd, bg, wm, bias,
                                    tanh_ah, u1, u2, p, q, bpack,
                                    (float*)d_out);
}
